// SWCAEncoderBlock_4647154614905
// MI455X (gfx1250) — matrix operand order repaired and checked by reading, not compiled, not run on hardware
//
#include <hip/hip_runtime.h>
#include <hip/hip_bf16.h>
#include <stdint.h>

typedef __bf16 bf16;
typedef bf16  v16bf __attribute__((ext_vector_type(16)));
typedef float v8f   __attribute__((ext_vector_type(8)));
typedef int   v2i   __attribute__((ext_vector_type(2)));

#define AS3 __attribute__((address_space(3)))
#define AS1 __attribute__((address_space(1)))

#if defined(__AMDGCN__) && __has_builtin(__builtin_amdgcn_global_load_async_to_lds_b64) && __has_builtin(__builtin_amdgcn_s_wait_asynccnt)
#define USE_ASYNC 1
/* arg0: global int2* (AS1, non-const), arg1: LDS int2* (AS3), then imm offset, cpol */
#define ASYNC_B64(g, l) __builtin_amdgcn_global_load_async_to_lds_b64((AS1 v2i*)(g), (AS3 v2i*)(l), 0, 0)
#else
#define USE_ASYNC 0
#endif

#define NUM_B   4
#define CQ      128
#define CH      256
#define HGT     256
#define WID     256
#define HW      (HGT*WID)
#define NPIX    (NUM_B*HW)
#define NHEADS  4
#define DH      64
#define WPIX    64
#define GROUPS  8
#define GN_CNT  ((CQ/GROUPS)*HW)   /* elements per (b,group) = 1048576 */
#define EPSF    1e-5f

/* ---------------- WMMA fragment loaders (wave32 layouts per CDNA5 ISA 7.12.2) ----- */
/* A (16x32 bf16): M = lane&15 ; element e -> K = (e/8)*16 + (lane>>4)*8 + (e%8)      */
__device__ inline v16bf load_afrag(const bf16* row, int hf) {
  union { uint2 u[4]; v16bf v; } f;
  f.u[0] = *(const uint2*)(row + hf*8);
  f.u[1] = *(const uint2*)(row + hf*8 + 4);
  f.u[2] = *(const uint2*)(row + 16 + hf*8);
  f.u[3] = *(const uint2*)(row + 16 + hf*8 + 4);
  return f.v;
}
/* B (32x16 bf16): N = lane&15 ; element e -> K = (e/8)*16 + (lane>>4)*8 + (e%8)      */
__device__ inline v16bf load_bfrag(const bf16* row, int hf) {
  union { uint2 u[4]; v16bf v; } f;
  f.u[0] = *(const uint2*)(row + hf*8);
  f.u[1] = *(const uint2*)(row + hf*8 + 4);
  f.u[2] = *(const uint2*)(row + 16 + hf*8);
  f.u[3] = *(const uint2*)(row + 16 + hf*8 + 4);
  return f.v;
}

/* xor-butterfly broadcast within the 16-lane half via ds_swizzle (group-of-32 mode) */
template<int XM>
__device__ inline float swz_xor(float v) {
  return __int_as_float(__builtin_amdgcn_ds_swizzle(__float_as_int(v), (XM << 10) | 0x1F));
}
__device__ inline float red_max16(float v) {
  v = fmaxf(v, swz_xor<1>(v)); v = fmaxf(v, swz_xor<2>(v));
  v = fmaxf(v, swz_xor<4>(v)); v = fmaxf(v, swz_xor<8>(v));
  return v;
}
__device__ inline float red_sum16(float v) {
  v += swz_xor<1>(v); v += swz_xor<2>(v);
  v += swz_xor<4>(v); v += swz_xor<8>(v);
  return v;
}

/* ---------------- small helpers ---------------- */
__global__ void k_zero_stats(float* stats) {
  int t = threadIdx.x;
  if (t < 192) stats[t] = 0.f;
}

__global__ void k_cvt_w(const float* __restrict__ w, bf16* __restrict__ o, int n) {
  int i = blockIdx.x * 256 + threadIdx.x;
  if (i < n) o[i] = (bf16)w[i];
}

/* ---------------- GroupNorm partial reduce: each (b,g) slab is contiguous --------- */
__global__ void k_gn_reduce(const float* __restrict__ x, float* __restrict__ stats) {
  int bg = blockIdx.x;           // 0..31  (b*GROUPS+g)
  int chunk = blockIdx.y;        // 0..63
  int tid = threadIdx.x;
  const float4* base = (const float4*)(x + (size_t)bg * GN_CNT + (size_t)chunk * (GN_CNT/64));
  float s = 0.f, ss = 0.f;
  for (int i = tid; i < (GN_CNT/64/4); i += 256) {
    float4 v = base[i];
    s  += v.x + v.y + v.z + v.w;
    ss += v.x*v.x + v.y*v.y + v.z*v.z + v.w*v.w;
  }
  __shared__ float sh1[256], sh2[256];
  sh1[tid] = s; sh2[tid] = ss;
  __syncthreads();
  for (int st = 128; st > 0; st >>= 1) {
    if (tid < st) { sh1[tid] += sh1[tid+st]; sh2[tid] += sh2[tid+st]; }
    __syncthreads();
  }
  if (tid == 0) {
    atomicAdd(&stats[bg*2+0], sh1[0]);
    atomicAdd(&stats[bg*2+1], sh2[0]);
  }
}

/* ---------------- GN normalize + NCHW -> [b*HW, C] bf16 transpose-pack ------------ */
__global__ void k_gn_norm_pack(const float* __restrict__ x, const float* __restrict__ stats,
                               const float* __restrict__ gw, const float* __restrict__ gb,
                               bf16* __restrict__ out, int C) {
  __shared__ float tile[32][33];
  int b = blockIdx.z, c0 = blockIdx.y * 32, p0 = blockIdx.x * 32;
  int tx = threadIdx.x, ty = threadIdx.y;
  int cpg = C / GROUPS;
  float inv_n = 1.f / ((float)cpg * (float)HW);
#pragma unroll
  for (int i = 0; i < 4; i++) {
    int c = c0 + ty + 8*i;
    int g = c / cpg;
    float s1 = stats[(b*GROUPS+g)*2+0];
    float s2 = stats[(b*GROUPS+g)*2+1];
    float mean = s1 * inv_n;
    float var  = s2 * inv_n - mean*mean;
    float rs   = rsqrtf(var + EPSF);
    float v = x[((size_t)b*C + c)*HW + p0 + tx];
    tile[ty+8*i][tx] = (v - mean) * rs * gw[c] + gb[c];
  }
  __syncthreads();
#pragma unroll
  for (int i = 0; i < 4; i++) {
    int p = p0 + ty + 8*i;
    int c = c0 + tx;
    out[((size_t)b*HW + p)*C + c] = (bf16)tile[tx][ty+8*i];
  }
}

/* ---------------- WMMA GEMM: Y[P,N] = X[P,K] * W[N,K]^T + bias, 4 epilogues ------- */
/* Double-buffered LDS with GLOBAL_LOAD_ASYNC_TO_LDS_B64 (ASYNCcnt) when available.  */
/* MODE 0: bf16 out ; 1: GELU+bf16 ; 2: +bias +x_q residual -> float NCHW            */
/* MODE 3: +bias, accumulate into float NCHW (d_out += )                             */
#define BM 128
#define BN 64
#define BK 32
#define KP 36   /* padded LDS row: 72B -> 8B aligned (async b64 + ds_load_b64 frags) */

template<int MODE>
__global__ void k_gemm(const bf16* __restrict__ X, const bf16* __restrict__ Wb,
                       const float* __restrict__ bias, int Kd, int Nd,
                       bf16* __restrict__ Ybf,
                       const float* __restrict__ xres, float* __restrict__ Yf) {
  __shared__ bf16 As[2][BM*KP];
  __shared__ bf16 Bs[2][BN*KP];
  const int tid  = threadIdx.x;
  const int lane = tid & 31, wave = tid >> 5;
  const int wm = wave & 3, wn = wave >> 2;
  const int hf = lane >> 4, l16 = lane & 15;
  const int m0 = blockIdx.x * BM;
  const int n0 = blockIdx.y * BN;
  const int ksteps = Kd / BK;

  /* issue one tile's staging (6 x 8B transfers per thread = 6 async ops per wave) */
  auto stage = [&](int kb, int buf) {
#pragma unroll
    for (int i = 0; i < 4; i++) {
      int idx = tid + i*256;             // 0..1023 : A tile 128x32
      int m = idx >> 3, seg = idx & 7;
      const bf16* g = X + (size_t)(m0+m)*Kd + kb*BK + seg*4;
      bf16* l = &As[buf][m*KP + seg*4];
#if USE_ASYNC
      ASYNC_B64(g, l);
#else
      *(uint2*)l = *(const uint2*)g;
#endif
    }
#pragma unroll
    for (int i = 0; i < 2; i++) {
      int idx = tid + i*256;             // 0..511 : B tile 64x32
      int n = idx >> 3, seg = idx & 7;
      const bf16* g = Wb + (size_t)(n0+n)*Kd + kb*BK + seg*4;
      bf16* l = &Bs[buf][n*KP + seg*4];
#if USE_ASYNC
      ASYNC_B64(g, l);
#else
      *(uint2*)l = *(const uint2*)g;
#endif
    }
  };

  v8f acc[2][2] = {};
  stage(0, 0);
  for (int kb = 0; kb < ksteps; kb++) {
    int cur = kb & 1;
    if (kb + 1 < ksteps) {
      stage(kb + 1, cur ^ 1);
#if USE_ASYNC
      __builtin_amdgcn_s_wait_asynccnt(6);   /* in-order: tile kb's 6 ops done */
#endif
    } else {
#if USE_ASYNC
      __builtin_amdgcn_s_wait_asynccnt(0);
#endif
    }
    __syncthreads();
    v16bf a0 = load_afrag(&As[cur][(wm*32 +  0 + l16)*KP], hf);
    v16bf a1 = load_afrag(&As[cur][(wm*32 + 16 + l16)*KP], hf);
    v16bf b0 = load_bfrag(&Bs[cur][(wn*32 +  0 + l16)*KP], hf);
    v16bf b1 = load_bfrag(&Bs[cur][(wn*32 + 16 + l16)*KP], hf);
    acc[0][0] = __builtin_amdgcn_wmma_f32_16x16x32_bf16(false, a0, false, b0, (short)0, acc[0][0], false, false);
    acc[0][1] = __builtin_amdgcn_wmma_f32_16x16x32_bf16(false, a0, false, b1, (short)0, acc[0][1], false, false);
    acc[1][0] = __builtin_amdgcn_wmma_f32_16x16x32_bf16(false, a1, false, b0, (short)0, acc[1][0], false, false);
    acc[1][1] = __builtin_amdgcn_wmma_f32_16x16x32_bf16(false, a1, false, b1, (short)0, acc[1][1], false, false);
    __syncthreads();
  }
  /* epilogue: C/D layout: row M = r + 8*(lane>>4), col N = lane&15 */
#pragma unroll
  for (int i = 0; i < 2; i++) {
#pragma unroll
    for (int j = 0; j < 2; j++) {
      int N = n0 + wn*32 + j*16 + l16;
      float bv = bias[N];
#pragma unroll
      for (int r = 0; r < 8; r++) {
        int M = m0 + wm*32 + i*16 + r + 8*hf;
        float v = acc[i][j][r] + bv;
        if (MODE == 0) {
          Ybf[(size_t)M*Nd + N] = (bf16)v;
        } else if (MODE == 1) {
          float t = 0.7978845608f * v * (1.f + 0.044715f * v * v);
          Ybf[(size_t)M*Nd + N] = (bf16)(0.5f * v * (1.f + tanhf(t)));
        } else {
          int b  = M >> 16;          /* HW = 65536 */
          int pp = M & 65535;
          size_t oidx = ((size_t)b*Nd + N)*HW + pp;
          if (MODE == 2) Yf[oidx] = v + xres[oidx];
          else           Yf[oidx] += v;
        }
      }
    }
  }
}

/* ---------------- fused window attention: one block per (window, head) ------------ */
#define DP 68   /* 136B LDS row stride: 8B aligned, 2-way conflicts */
__global__ void k_attn(const bf16* __restrict__ Q, const bf16* __restrict__ Kt,
                       const bf16* __restrict__ V, bf16* __restrict__ A) {
  __shared__ bf16 qs[WPIX*DP];
  __shared__ bf16 ks[WPIX*DP];
  __shared__ bf16 vt[DH*DP];     /* V transposed: [d][key] */
  __shared__ bf16 ps[WPIX*DP];   /* softmax probs */
  int wi = blockIdx.x, head = blockIdx.y;
  int b   = wi >> 10;
  int wl  = wi & 1023;
  int whi = wl >> 5, wwi = wl & 31;
  int tid = threadIdx.x, lane = tid & 31, wave = tid >> 5;
  int hf = lane >> 4, l16 = lane & 15;

  /* stage Q,K rows (64 x 64 bf16 each) — async when available */
  for (int idx = tid; idx < 1024; idx += 128) {
    int l = idx >> 4, seg = idx & 15;
    int wr = l >> 3, wc = l & 7;
    size_t p    = (size_t)b*HW + (size_t)(whi*8 + wr)*WID + wwi*8 + wc;
    size_t goff = p*CH + head*DH + seg*4;
#if USE_ASYNC
    ASYNC_B64(Q  + goff, &qs[l*DP + seg*4]);
    ASYNC_B64(Kt + goff, &ks[l*DP + seg*4]);
#else
    *(uint2*)(&qs[l*DP + seg*4]) = *(const uint2*)(Q  + goff);
    *(uint2*)(&ks[l*DP + seg*4]) = *(const uint2*)(Kt + goff);
#endif
  }
  /* stage V transposed (scatter -> must go through VGPRs) */
  for (int idx = tid; idx < 1024; idx += 128) {
    int l = idx >> 4, seg = idx & 15;
    int wr = l >> 3, wc = l & 7;
    size_t p    = (size_t)b*HW + (size_t)(whi*8 + wr)*WID + wwi*8 + wc;
    size_t goff = p*CH + head*DH + seg*4;
    union { uint2 u; bf16 h[4]; } t;
    t.u = *(const uint2*)(V + goff);
#pragma unroll
    for (int q2 = 0; q2 < 4; q2++) vt[(seg*4 + q2)*DP + l] = t.h[q2];
  }
#if USE_ASYNC
  __builtin_amdgcn_s_wait_asynccnt(0);
#endif
  __syncthreads();

  /* S = q * k^T  (wave w owns query rows 16w..16w+15) */
  v8f sacc[4] = {};
  for (int kk = 0; kk < 2; kk++) {
    v16bf af = load_afrag(&qs[(wave*16 + l16)*DP + kk*32], hf);
#pragma unroll
    for (int t = 0; t < 4; t++) {
      v16bf bfv = load_bfrag(&ks[(t*16 + l16)*DP + kk*32], hf);
      sacc[t] = __builtin_amdgcn_wmma_f32_16x16x32_bf16(false, af, false, bfv, (short)0, sacc[t], false, false);
    }
  }
  /* softmax over keys; row = wave*16 + r + 8*hf, cols spread over (t, l16) */
  const float scale = 0.125f;   /* dh=64 -> 64^-0.5 */
#pragma unroll
  for (int r = 0; r < 8; r++) {
    float mx = -3.0e38f;
#pragma unroll
    for (int t = 0; t < 4; t++) { sacc[t][r] *= scale; mx = fmaxf(mx, sacc[t][r]); }
    mx = red_max16(mx);
    float sm = 0.f;
#pragma unroll
    for (int t = 0; t < 4; t++) { float e = __expf(sacc[t][r] - mx); sacc[t][r] = e; sm += e; }
    sm = red_sum16(sm);
    float inv = 1.f / sm;
#pragma unroll
    for (int t = 0; t < 4; t++) sacc[t][r] *= inv;
  }
  /* C-frag -> A-frag relayout through LDS (wave-private rows; DS is in-order) */
#pragma unroll
  for (int t = 0; t < 4; t++)
#pragma unroll
    for (int r = 0; r < 8; r++)
      ps[(wave*16 + r + 8*hf)*DP + t*16 + l16] = (bf16)sacc[t][r];

  /* O = P * V */
  v8f oacc[4] = {};
  for (int kk = 0; kk < 2; kk++) {
    v16bf af = load_afrag(&ps[(wave*16 + l16)*DP + kk*32], hf);
#pragma unroll
    for (int t = 0; t < 4; t++) {
      v16bf bfv = load_bfrag(&vt[(t*16 + l16)*DP + kk*32], hf);
      oacc[t] = __builtin_amdgcn_wmma_f32_16x16x32_bf16(false, af, false, bfv, (short)0, oacc[t], false, false);
    }
  }
  /* scatter O back to [pixel, head*64+d] bf16 */
#pragma unroll
  for (int t = 0; t < 4; t++)
#pragma unroll
    for (int r = 0; r < 8; r++) {
      int m = wave*16 + r + 8*hf;
      int wr = m >> 3, wc = m & 7;
      size_t p = (size_t)b*HW + (size_t)(whi*8 + wr)*WID + wwi*8 + wc;
      A[p*CH + head*DH + t*16 + l16] = (bf16)oacc[t][r];
    }
}

/* ---------------- launch ---------------- */
extern "C" void kernel_launch(void* const* d_in, const int* in_sizes, int n_in,
                              void* d_out, int out_size, void* d_ws, size_t ws_size,
                              hipStream_t stream) {
  (void)in_sizes; (void)n_in; (void)out_size; (void)ws_size;
  const float* x_q    = (const float*)d_in[0];
  const float* x_kv   = (const float*)d_in[1];
  const float* gn_q_w = (const float*)d_in[2];
  const float* gn_q_b = (const float*)d_in[3];
  const float* gn_kv_w= (const float*)d_in[4];
  const float* gn_kv_b= (const float*)d_in[5];
  const float* gn2_w  = (const float*)d_in[6];
  const float* gn2_b  = (const float*)d_in[7];
  const float* bq = (const float*)d_in[9];
  const float* bk = (const float*)d_in[11];
  const float* bv = (const float*)d_in[13];
  const float* bp = (const float*)d_in[15];
  const float* b1 = (const float*)d_in[17];
  const float* b2 = (const float*)d_in[19];
  float* out = (float*)d_out;

  /* ws layout (needs ~641 MiB) */
  char* ws = (char*)d_ws;
  const size_t SZ_QN = (size_t)NPIX * CQ * 2;   /* 64 MiB */
  const size_t SZ_CH = (size_t)NPIX * CH * 2;   /* 128 MiB */
  float* stats = (float*)(ws + 0);              /* 3 slots x 64 floats */
  bf16*  wbf   = (bf16*)(ws + 4096);            /* 6 x 32768 bf16 */
  bf16*  qn    = (bf16*)(ws + (1u<<20));
  bf16*  kvn   = (bf16*)(ws + (1u<<20) + SZ_QN);
  bf16*  Qb    = (bf16*)(ws + (1u<<20) + 2*SZ_QN);
  bf16*  Kb    = (bf16*)(ws + (1u<<20) + 2*SZ_QN + SZ_CH);
  bf16*  Vb    = (bf16*)(ws + (1u<<20) + 2*SZ_QN + 2*SZ_CH);
  bf16*  Ab    = (bf16*)(ws + (1u<<20) + 2*SZ_QN + 3*SZ_CH);
  bf16*  hn = qn;   /* reuse after attention consumed qn */
  bf16*  h1 = Qb;   /* reuse after attention consumed Qb */

  k_zero_stats<<<1, 256, 0, stream>>>(stats);

  k_cvt_w<<<128, 256, 0, stream>>>((const float*)d_in[8],  wbf + 0,      32768);
  k_cvt_w<<<128, 256, 0, stream>>>((const float*)d_in[10], wbf + 32768,  32768);
  k_cvt_w<<<128, 256, 0, stream>>>((const float*)d_in[12], wbf + 65536,  32768);
  k_cvt_w<<<128, 256, 0, stream>>>((const float*)d_in[14], wbf + 98304,  32768);
  k_cvt_w<<<128, 256, 0, stream>>>((const float*)d_in[16], wbf + 131072, 32768);
  k_cvt_w<<<128, 256, 0, stream>>>((const float*)d_in[18], wbf + 163840, 32768);

  dim3 gr(32, 64);
  k_gn_reduce<<<gr, 256, 0, stream>>>(x_q,  stats + 0);
  k_gn_reduce<<<gr, 256, 0, stream>>>(x_kv, stats + 64);

  dim3 gnp(HW/32, CQ/32, NUM_B), bnp(32, 8);
  k_gn_norm_pack<<<gnp, bnp, 0, stream>>>(x_q,  stats + 0,  gn_q_w,  gn_q_b,  qn,  CQ);
  k_gn_norm_pack<<<gnp, bnp, 0, stream>>>(x_kv, stats + 64, gn_kv_w, gn_kv_b, kvn, CQ);

  dim3 gq(NPIX/BM, CH/BN);     /* 2048 x 4 */
  k_gemm<0><<<gq, 256, 0, stream>>>(qn,  wbf + 0,     bq, CQ, CH, Qb, nullptr, nullptr);
  k_gemm<0><<<gq, 256, 0, stream>>>(kvn, wbf + 32768, bk, CQ, CH, Kb, nullptr, nullptr);
  k_gemm<0><<<gq, 256, 0, stream>>>(kvn, wbf + 65536, bv, CQ, CH, Vb, nullptr, nullptr);

  dim3 ga(NUM_B * 1024, NHEADS);
  k_attn<<<ga, 128, 0, stream>>>(Qb, Kb, Vb, Ab);

  dim3 gp(NPIX/BM, CQ/BN);     /* 2048 x 2 */
  k_gemm<2><<<gp, 256, 0, stream>>>(Ab, wbf + 98304, bp, CH, CQ, nullptr, x_q, out);

  k_gn_reduce<<<gr, 256, 0, stream>>>((const float*)out, stats + 128);
  k_gn_norm_pack<<<gnp, bnp, 0, stream>>>((const float*)out, stats + 128, gn2_w, gn2_b, hn, CQ);

  k_gemm<1><<<gq, 256, 0, stream>>>(hn, wbf + 131072, b1, CQ, CH, h1, nullptr, nullptr);
  k_gemm<3><<<gp, 256, 0, stream>>>(h1, wbf + 163840, b2, CH, CQ, nullptr, nullptr, out);
}
